// MACELayer_81200651698610
// MI455X (gfx1250) — matrix operand order repaired and checked by reading, not compiled, not run on hardware
//
#include <hip/hip_runtime.h>

typedef __attribute__((ext_vector_type(16))) _Float16 v16h;
typedef __attribute__((ext_vector_type(8)))  float    v8f;

#define N_NODES 20000
#define N_EDGES 320000
#define C_CH    64
#define B_RAD   8
#define H_HID   64
#define NPATH   5
#define INV_AVG 0.0625f   // 1/16 avg neighbors

// ---------------- WMMA helpers (gfx1250, wave32, 16x16x32 f16 -> f32) -------

__device__ __forceinline__ v8f wmma_f16(v16h a, v16h b, v8f c) {
    // 8 args: (neg_a, A, neg_b, B, c_mod, C, reuse_a, reuse_b)
    return __builtin_amdgcn_wmma_f32_16x16x32_f16(false, a, false, b, (short)0, c,
                                                  false, false);
}

// A fragment, contiguous (es==1) source: 16x32 tile of row-major f32 matrix.
// Per lane the 16 elements are two contiguous 8-float runs -> 4x float4 loads.
__device__ __forceinline__ v16h load_a_f32_c(const float* __restrict__ X, int ld,
                                             int m0, int k0, int lane, float scale) {
    int m  = m0 + (lane & 15);
    int kb = (lane & 16) ? 8 : 0;
    const float* base = X + m * ld + k0 + kb;
    float4 p0 = *reinterpret_cast<const float4*>(base + 0);
    float4 p1 = *reinterpret_cast<const float4*>(base + 4);
    float4 p2 = *reinterpret_cast<const float4*>(base + 16);
    float4 p3 = *reinterpret_cast<const float4*>(base + 20);
    v16h a;
    a[0]  = (_Float16)(p0.x * scale); a[1]  = (_Float16)(p0.y * scale);
    a[2]  = (_Float16)(p0.z * scale); a[3]  = (_Float16)(p0.w * scale);
    a[4]  = (_Float16)(p1.x * scale); a[5]  = (_Float16)(p1.y * scale);
    a[6]  = (_Float16)(p1.z * scale); a[7]  = (_Float16)(p1.w * scale);
    a[8]  = (_Float16)(p2.x * scale); a[9]  = (_Float16)(p2.y * scale);
    a[10] = (_Float16)(p2.z * scale); a[11] = (_Float16)(p2.w * scale);
    a[12] = (_Float16)(p3.x * scale); a[13] = (_Float16)(p3.y * scale);
    a[14] = (_Float16)(p3.z * scale); a[15] = (_Float16)(p3.w * scale);
    return a;
}

// A fragment, strided source (es==3, vector components interleaved [.,C,3]).
__device__ __forceinline__ v16h load_a_f32_s3(const float* __restrict__ X, int ld,
                                              int m0, int k0, int lane, float scale) {
    int m  = m0 + (lane & 15);
    int kb = (lane & 16) ? 8 : 0;
    int base = (m * ld + k0 + kb) * 3;
    v16h a;
#pragma unroll
    for (int e = 0; e < 16; ++e) {
        int K = (e < 8) ? e : (e + 8);       // contiguous runs at +0 and +16
        a[e] = (_Float16)(X[base + K * 3] * scale);
    }
    return a;
}

// B fragment: 32x16 tile of row-major f32 weight W [K x ld], origin (k0, n0).
// Lane = column; lanes 0-15 hold K=0..7,16..23, lanes 16-31 K=8..15,24..31.
__device__ __forceinline__ v16h load_b_f32(const float* __restrict__ W, int ld,
                                           int k0, int n0, int lane) {
    int n  = n0 + (lane & 15);
    int kb = (lane & 16) ? 8 : 0;
    int base = (k0 + kb) * ld + n;
    v16h b;
#pragma unroll
    for (int e = 0; e < 16; ++e)
        b[e] = (_Float16)W[base + ((e < 8) ? e : (e + 8)) * ld];
    return b;
}

// Same, but from an f16 matrix already staged in LDS.
__device__ __forceinline__ v16h load_b_lds_h(const _Float16* __restrict__ W, int ld,
                                             int k0, int n0, int lane) {
    int n  = n0 + (lane & 15);
    int kb = (lane & 16) ? 8 : 0;
    int base = (k0 + kb) * ld + n;
    v16h b;
#pragma unroll
    for (int e = 0; e < 16; ++e)
        b[e] = W[base + ((e < 8) ? e : (e + 8)) * ld];
    return b;
}

// D fragment store: element r -> row m0 + r + 8*laneHi, col n0 + (lane&15).
__device__ __forceinline__ void store_d(const v8f& d, float* __restrict__ Y, int ld,
                                        int es, int m0, int n0, int lane) {
    int n  = n0 + (lane & 15);
    int mh = (lane & 16) ? 8 : 0;
    int base = ((m0 + mh) * ld + n) * es;
#pragma unroll
    for (int r = 0; r < 8; ++r)
        Y[base + r * ld * es] = d[r];
}

// ---------------- Kernel 0: zero the scatter accumulators -------------------

__global__ void k_zero(float4* __restrict__ p, unsigned n4) {
    unsigned i = blockIdx.x * blockDim.x + threadIdx.x;
    if (i < n4) p[i] = make_float4(0.f, 0.f, 0.f, 0.f);
}

// ---------------- Kernel 1: linear_up (s and 3 components of v) -------------

__global__ __launch_bounds__(256) void k_up(const float* __restrict__ node_s,
                                            const float* __restrict__ node_v,
                                            const float* __restrict__ Wus,
                                            const float* __restrict__ Wuv,
                                            float* __restrict__ s_up,
                                            float* __restrict__ v_up) {
    int lane = threadIdx.x & 31;
    int wave = threadIdx.x >> 5;
    int m0   = blockIdx.x * 16;
#pragma unroll
    for (int i = 0; i < 2; ++i) {
        int job = wave * 2 + i;
        int mat = job >> 2;          // 0: scalar, 1..3: vector component d
        int n0  = (job & 3) * 16;
        v8f acc = {};
        if (mat == 0) {
            acc = wmma_f16(load_a_f32_c(node_s, C_CH, m0, 0,  lane, 1.0f),
                           load_b_f32(Wus, C_CH, 0,  n0, lane), acc);
            acc = wmma_f16(load_a_f32_c(node_s, C_CH, m0, 32, lane, 1.0f),
                           load_b_f32(Wus, C_CH, 32, n0, lane), acc);
            store_d(acc, s_up, C_CH, 1, m0, n0, lane);
        } else {
            int d = mat - 1;
            acc = wmma_f16(load_a_f32_s3(node_v + d, C_CH, m0, 0,  lane, 1.0f),
                           load_b_f32(Wuv, C_CH, 0,  n0, lane), acc);
            acc = wmma_f16(load_a_f32_s3(node_v + d, C_CH, m0, 32, lane, 1.0f),
                           load_b_f32(Wuv, C_CH, 32, n0, lane), acc);
            store_d(acc, v_up + d, C_CH, 3, m0, n0, lane);
        }
    }
}

// ---------------- Kernel 2: fused radial-MLP + messages + scatter -----------
// One 16-edge tile per wave; wts [16 x 320] lives only in D fragments.

__global__ __launch_bounds__(256) void k_edge(const float* __restrict__ vectors,
                                              const float* __restrict__ radial,
                                              const int*   __restrict__ senders,
                                              const int*   __restrict__ receivers,
                                              const float* __restrict__ Wr1,
                                              const float* __restrict__ br1,
                                              const float* __restrict__ Wr2,
                                              const float* __restrict__ br2,
                                              const float* __restrict__ s_up,
                                              const float* __restrict__ v_up,
                                              float* __restrict__ agg_s,
                                              float* __restrict__ agg_v) {
    __shared__ __align__(16) _Float16 Wr2h[H_HID * NPATH * C_CH];  // 40 KB
    __shared__ __align__(16) float    Wr1s[B_RAD * H_HID];
    __shared__ float                  br1s[H_HID];

    for (int i = threadIdx.x; i < H_HID * NPATH * C_CH; i += 256)
        Wr2h[i] = (_Float16)Wr2[i];
    for (int i = threadIdx.x; i < B_RAD * H_HID; i += 256) Wr1s[i] = Wr1[i];
    if (threadIdx.x < H_HID) br1s[threadIdx.x] = br1[threadIdx.x];
    __syncthreads();

    int lane = threadIdx.x & 31;
    int wave = threadIdx.x >> 5;
    int tile = blockIdx.x * 8 + wave;
    int e0   = tile * 16;
    int hi   = (lane >> 4) & 1;
    int ei_a = e0 + (lane & 15);

    // radial layer 1 -> silu -> directly into A-fragment layout (K = 0..63)
    float4 r0 = *reinterpret_cast<const float4*>(radial + ei_a * B_RAD);
    float4 r1 = *reinterpret_cast<const float4*>(radial + ei_a * B_RAD + 4);
    float re[B_RAD] = {r0.x, r0.y, r0.z, r0.w, r1.x, r1.y, r1.z, r1.w};

    int kb = hi ? 8 : 0;
    v16h a0, a1;
#pragma unroll
    for (int e = 0; e < 16; ++e) {
        int K = ((e < 8) ? e : (e + 8)) + kb;
        float acc0 = br1s[K];
        float acc1 = br1s[K + 32];
#pragma unroll
        for (int j = 0; j < B_RAD; ++j) {
            acc0 += re[j] * Wr1s[j * H_HID + K];
            acc1 += re[j] * Wr1s[j * H_HID + K + 32];
        }
        acc0 = acc0 / (1.0f + __expf(-acc0));   // silu
        acc1 = acc1 / (1.0f + __expf(-acc1));
        a0[e] = (_Float16)acc0;
        a1[e] = (_Float16)acc1;
    }

    // hoist per-edge geometry/topology for this lane's 8 D-fragment rows
    float Yx[8], Yy[8], Yz[8];
    int   sbase[8], rbase[8];
#pragma unroll
    for (int r = 0; r < 8; ++r) {
        int ei = e0 + hi * 8 + r;
        const float* vp = vectors + ei * 3;
        float vx = vp[0], vy = vp[1], vz = vp[2];
        float rinv = 1.0f / (sqrtf(vx * vx + vy * vy + vz * vz) + 1e-9f);
        Yx[r] = vx * rinv; Yy[r] = vy * rinv; Yz[r] = vz * rinv;
        sbase[r] = senders[ei]   * C_CH;
        rbase[r] = receivers[ei] * C_CH;
    }

    for (int c16 = 0; c16 < 4; ++c16) {
        v8f wfrag[NPATH];
#pragma unroll
        for (int p = 0; p < NPATH; ++p) {
            int n0 = p * C_CH + c16 * 16;
            float bias = br2[n0 + (lane & 15)];
            v8f acc = {bias, bias, bias, bias, bias, bias, bias, bias};
            acc = wmma_f16(a0, load_b_lds_h(Wr2h, NPATH * C_CH, 0,  n0, lane), acc);
            acc = wmma_f16(a1, load_b_lds_h(Wr2h, NPATH * C_CH, 32, n0, lane), acc);
            wfrag[p] = acc;
        }
        int c = c16 * 16 + (lane & 15);
#pragma unroll
        for (int r = 0; r < 8; ++r) {
            float w0 = wfrag[0][r], w1 = wfrag[1][r], w2 = wfrag[2][r];
            float w3 = wfrag[3][r], w4 = wfrag[4][r];
            int sb = sbase[r] + c;
            float ss  = s_up[sb];
            const float* vp = v_up + sb * 3;     // 3 contiguous floats (b96)
            float sv0 = vp[0], sv1 = vp[1], sv2 = vp[2];
            float dot = sv0 * Yx[r] + sv1 * Yy[r] + sv2 * Yz[r];
            float cr0 = sv1 * Yz[r] - sv2 * Yy[r];
            float cr1 = sv2 * Yx[r] - sv0 * Yz[r];
            float cr2 = sv0 * Yy[r] - sv1 * Yx[r];
            float ms  = w0 * ss + w3 * dot;
            float wy  = w1 * ss;
            float mv0 = wy * Yx[r] + w2 * sv0 + w4 * cr0;
            float mv1 = wy * Yy[r] + w2 * sv1 + w4 * cr1;
            float mv2 = wy * Yz[r] + w2 * sv2 + w4 * cr2;
            int rb = rbase[r] + c;
            atomicAdd(&agg_s[rb], ms);
            atomicAdd(&agg_v[rb * 3 + 0], mv0);
            atomicAdd(&agg_v[rb * 3 + 1], mv1);
            atomicAdd(&agg_v[rb * 3 + 2], mv2);
        }
    }
}

// ---------------- Kernel 3: linear_down -> selector -> symcontract ----------
//                 -> post linear -> readout   (fused per 16-node tile)

__global__ __launch_bounds__(128) void k_node(const float* __restrict__ agg_s,
                                              const float* __restrict__ agg_v,
                                              const float* __restrict__ Wds,
                                              const float* __restrict__ Wdv,
                                              const int*   __restrict__ specie,
                                              const float* __restrict__ Wsel_s,
                                              const float* __restrict__ Wsel_v,
                                              const float* __restrict__ Wsym_s,
                                              const float* __restrict__ Wsym_v,
                                              const float* __restrict__ Wps,
                                              const float* __restrict__ Wpv,
                                              const float* __restrict__ Wread,
                                              float* __restrict__ out_read,
                                              float* __restrict__ out_s,
                                              float* __restrict__ out_v) {
    __shared__ __align__(16) float sA[16 * 64];
    __shared__ __align__(16) float vA[16 * 64 * 3];
    __shared__ __align__(16) float sB[16 * 64];
    __shared__ __align__(16) float vB[16 * 64 * 3];
    __shared__ __align__(16) float sC[16 * 64];

    int lane = threadIdx.x & 31;
    int wave = threadIdx.x >> 5;
    int m0   = blockIdx.x * 16;

    // Stage 1: linear_down via WMMA, 1/AVG_NEIGH folded into the A load.
#pragma unroll
    for (int i = 0; i < 4; ++i) {
        int job = wave * 4 + i;
        int mat = job >> 2;
        int n0  = (job & 3) * 16;
        v8f acc = {};
        if (mat == 0) {
            acc = wmma_f16(load_a_f32_c(agg_s, C_CH, m0, 0,  lane, INV_AVG),
                           load_b_f32(Wds, C_CH, 0,  n0, lane), acc);
            acc = wmma_f16(load_a_f32_c(agg_s, C_CH, m0, 32, lane, INV_AVG),
                           load_b_f32(Wds, C_CH, 32, n0, lane), acc);
            store_d(acc, sA, C_CH, 1, 0, n0, lane);
        } else {
            int d = mat - 1;
            acc = wmma_f16(load_a_f32_s3(agg_v + d, C_CH, m0, 0,  lane, INV_AVG),
                           load_b_f32(Wdv, C_CH, 0,  n0, lane), acc);
            acc = wmma_f16(load_a_f32_s3(agg_v + d, C_CH, m0, 32, lane, INV_AVG),
                           load_b_f32(Wdv, C_CH, 32, n0, lane), acc);
            store_d(acc, vA + d, C_CH, 3, 0, n0, lane);
        }
    }
    __syncthreads();

    // Stage 2: species selector mix + symmetric contraction, elementwise (n,e)
#pragma unroll
    for (int i = 0; i < 8; ++i) {
        int pair = threadIdx.x + i * 128;
        int n = pair >> 6, e = pair & 63;
        int sp = specie[m0 + n];
        int wb = sp * 64 * 64 + e;
        float ssel = 0.f, v0 = 0.f, v1 = 0.f, v2 = 0.f;
        for (int ci = 0; ci < 64; ++ci) {
            float wsc = Wsel_s[wb + ci * 64];
            float wvc = Wsel_v[wb + ci * 64];
            ssel += sA[n * 64 + ci] * wsc;
            v0   += vA[(n * 64 + ci) * 3 + 0] * wvc;
            v1   += vA[(n * 64 + ci) * 3 + 1] * wvc;
            v2   += vA[(n * 64 + ci) * 3 + 2] * wvc;
        }
        float s2 = ssel * ssel;
        float vv = v0 * v0 + v1 * v1 + v2 * v2;
        int qsb = sp * 5 * 64 + e;
        int qvb = sp * 4 * 64 + e;
        float so = Wsym_s[qsb] * ssel + Wsym_s[qsb + 64] * s2
                 + Wsym_s[qsb + 128] * (ssel * s2)
                 + Wsym_s[qsb + 192] * vv + Wsym_s[qsb + 256] * (ssel * vv);
        float fv = Wsym_v[qvb] + Wsym_v[qvb + 64] * ssel
                 + Wsym_v[qvb + 128] * s2 + Wsym_v[qvb + 192] * vv;
        sB[n * 64 + e] = so;
        vB[(n * 64 + e) * 3 + 0] = v0 * fv;
        vB[(n * 64 + e) * 3 + 1] = v1 * fv;
        vB[(n * 64 + e) * 3 + 2] = v2 * fv;
    }
    __syncthreads();

    // Stage 3: post linear via WMMA; write final s and v.
#pragma unroll
    for (int i = 0; i < 4; ++i) {
        int job = wave * 4 + i;
        int mat = job >> 2;
        int n0  = (job & 3) * 16;
        v8f acc = {};
        if (mat == 0) {
            acc = wmma_f16(load_a_f32_c(sB, C_CH, 0, 0,  lane, 1.0f),
                           load_b_f32(Wps, C_CH, 0,  n0, lane), acc);
            acc = wmma_f16(load_a_f32_c(sB, C_CH, 0, 32, lane, 1.0f),
                           load_b_f32(Wps, C_CH, 32, n0, lane), acc);
            store_d(acc, sC, C_CH, 1, 0, n0, lane);          // keep for readout
            store_d(acc, out_s, C_CH, 1, m0, n0, lane);
        } else {
            int d = mat - 1;
            acc = wmma_f16(load_a_f32_s3(vB + d, C_CH, 0, 0,  lane, 1.0f),
                           load_b_f32(Wpv, C_CH, 0,  n0, lane), acc);
            acc = wmma_f16(load_a_f32_s3(vB + d, C_CH, 0, 32, lane, 1.0f),
                           load_b_f32(Wpv, C_CH, 32, n0, lane), acc);
            store_d(acc, out_v + d, C_CH, 3, m0, n0, lane);
        }
    }
    __syncthreads();

    // Stage 4: readout (0e head)
    if (threadIdx.x < 16) {
        int n = threadIdx.x;
        float acc = 0.f;
        for (int e = 0; e < 64; ++e) acc += sC[n * 64 + e] * Wread[e];
        out_read[m0 + n] = acc;
    }
}

// ---------------- Host-side launch ------------------------------------------

extern "C" void kernel_launch(void* const* d_in, const int* in_sizes, int n_in,
                              void* d_out, int out_size, void* d_ws, size_t ws_size,
                              hipStream_t stream) {
    const float* vectors   = (const float*)d_in[0];
    const float* node_s    = (const float*)d_in[1];
    const float* node_v    = (const float*)d_in[2];
    const float* radial    = (const float*)d_in[3];
    const int*   specie    = (const int*)  d_in[4];
    const int*   senders   = (const int*)  d_in[5];
    const int*   receivers = (const int*)  d_in[6];
    const float* Wus   = (const float*)d_in[7];
    const float* Wuv   = (const float*)d_in[8];
    const float* Wr1   = (const float*)d_in[9];
    const float* br1   = (const float*)d_in[10];
    const float* Wr2   = (const float*)d_in[11];
    const float* br2   = (const float*)d_in[12];
    const float* Wds   = (const float*)d_in[13];
    const float* Wdv   = (const float*)d_in[14];
    const float* Wsels = (const float*)d_in[15];
    const float* Wselv = (const float*)d_in[16];
    const float* Wsyms = (const float*)d_in[17];
    const float* Wsymv = (const float*)d_in[18];
    const float* Wps   = (const float*)d_in[19];
    const float* Wpv   = (const float*)d_in[20];
    const float* Wread = (const float*)d_in[21];

    float* ws    = (float*)d_ws;
    float* s_up  = ws;                                     // [N,C]
    float* v_up  = s_up  + (size_t)N_NODES * C_CH;         // [N,C,3]
    float* agg_s = v_up  + (size_t)N_NODES * C_CH * 3;     // [N,C]
    float* agg_v = agg_s + (size_t)N_NODES * C_CH;         // [N,C,3]

    float* out      = (float*)d_out;
    float* out_read = out;                                 // [N,1,1]
    float* out_s    = out + N_NODES;                       // [N,C]
    float* out_v    = out_s + (size_t)N_NODES * C_CH;      // [N,C,3]

    unsigned n4 = (unsigned)(N_NODES * C_CH * 4 / 4);      // agg_s + agg_v, float4s
    k_zero<<<(n4 + 255) / 256, 256, 0, stream>>>((float4*)agg_s, n4);

    k_up<<<N_NODES / 16, 256, 0, stream>>>(node_s, node_v, Wus, Wuv, s_up, v_up);

    k_edge<<<N_EDGES / 16 / 8, 256, 0, stream>>>(vectors, radial, senders, receivers,
                                                 Wr1, br1, Wr2, br2,
                                                 s_up, v_up, agg_s, agg_v);

    k_node<<<N_NODES / 16, 128, 0, stream>>>(agg_s, agg_v, Wds, Wdv, specie,
                                             Wsels, Wselv, Wsyms, Wsymv,
                                             Wps, Wpv, Wread,
                                             out_read, out_s, out_v);
}
